// GridNet_50208167690289
// MI455X (gfx1250) — matrix operand order repaired and checked by reading, not compiled, not run on hardware
//
#include <hip/hip_runtime.h>

#define GHN 256
#define GWN 256
#define FNUM 32
#define WAVES 8

typedef __attribute__((ext_vector_type(16))) _Float16 v16h;
typedef __attribute__((ext_vector_type(8)))  float    v8f;

// slope 0.01 > 0  =>  leaky_relu(x) == max(x, 0.01*x)   (2 VALU, dual-issue friendly)
__device__ __forceinline__ float leakyf(float x) { return fmaxf(x, 0.01f * x); }
__device__ __forceinline__ float sigmf(float x) { return 1.f / (1.f + expf(-x)); }

// A fragment (16x32, f16). lane m = row; VGPR v holds K pair per ISA A-layout.
__device__ __forceinline__ v16h load_A_frag(const _Float16* t, int stride, int kbase, int lane) {
  const int m = lane & 15, h = lane >> 4;
  v16h a;
#pragma unroll
  for (int v = 0; v < 8; ++v) {
    const int k = (v < 4 ? 2 * v : 16 + 2 * (v - 4)) + h * 8 + kbase;
    a[2 * v]     = t[m * stride + k];
    a[2 * v + 1] = t[m * stride + k + 1];
  }
  return a;
}

// B fragment (32x16, f16). lane n = column; K order matches A fragment (8h+s, 16+8h+s-8).
__device__ __forceinline__ v16h load_B_frag(const _Float16* w, int stride, int k0, int n0, int lane) {
  const int nn = lane & 15, h = lane >> 4;
  v16h b;
#pragma unroll
  for (int v = 0; v < 8; ++v) {
    const int k = k0 + (v < 4 ? 2 * v : 16 + 2 * (v - 4)) + 8 * h;
    b[2 * v]     = w[k * stride + n0 + nn];
    b[2 * v + 1] = w[(k + 1) * stride + n0 + nn];
  }
  return b;
}

__global__ __launch_bounds__(256) void gridnet_kernel(
    const float* __restrict__ pos, const float* __restrict__ dir,
    const float* __restrict__ pos_grid, const float* __restrict__ dir_grid,
    const float* __restrict__ ew1, const float* __restrict__ eb1,
    const float* __restrict__ ew2, const float* __restrict__ eb2,
    const float* __restrict__ w1, const float* __restrict__ b1,
    const float* __restrict__ w2, const float* __restrict__ b2,
    const float* __restrict__ w3, const float* __restrict__ b3,
    const float* __restrict__ w4, const float* __restrict__ b4,
    float* __restrict__ out, int n, int ntiles)
{
  __shared__ _Float16 sW1[64 * 64];        // FC1 weights f16
  __shared__ _Float16 sW2[64 * 16];        // FC2 weights f16
  __shared__ _Float16 sW3[32 * 16];        // FC3 weights, zero-padded K->32, N->16
  __shared__ float sW4[24], sB1[64], sB2[16], sB3[8], sB4[3];
  __shared__ float sE[27];                 // ew1(8) eb1(4) ew2(12) eb2(3)
  __shared__ _Float16 sX[WAVES][16 * 64];  // per-wave activation tile (feat / act1)
  __shared__ _Float16 sX3[WAVES][16 * 32]; // per-wave FC3 input tile (K zero-pad static)
  __shared__ float sY[WAVES][16 * 8];      // per-wave FC3 output transpose buffer
  __shared__ float sO[WAVES][48];          // per-wave output staging (16 pts x 3)

  const int tid = threadIdx.x;
  for (int i = tid; i < 64 * 64; i += 256) sW1[i] = (_Float16)w1[i];
  for (int i = tid; i < 64 * 16; i += 256) sW2[i] = (_Float16)w2[i];
  for (int i = tid; i < 32 * 16; i += 256) {
    int k = i >> 4, nn = i & 15;
    sW3[i] = (k < 16 && nn < 8) ? (_Float16)w3[k * 8 + nn] : (_Float16)0.f;
  }
  if (tid < 24) sW4[tid] = w4[tid];
  if (tid < 64) sB1[tid] = b1[tid];
  if (tid < 16) sB2[tid] = b2[tid];
  if (tid < 8)  sB3[tid] = b3[tid];
  if (tid < 3)  sB4[tid] = b4[tid];
  if (tid < 8)       sE[tid] = ew1[tid];
  else if (tid < 12) sE[tid] = eb1[tid - 8];
  else if (tid < 24) sE[tid] = ew2[tid - 12];
  else if (tid < 27) sE[tid] = eb2[tid - 24];
  __syncthreads();

  const int wave = tid >> 5;
  const int lane = tid & 31;
  const int m    = lane & 15;
  const int half = lane >> 4;

  // Hoist all B fragments into VGPRs (constant across tiles).
  v16h B1f[4][2], B2f[2], B3f;
#pragma unroll
  for (int nt = 0; nt < 4; ++nt)
#pragma unroll
    for (int kc = 0; kc < 2; ++kc)
      B1f[nt][kc] = load_B_frag(sW1, 64, kc * 32, nt * 16, lane);
#pragma unroll
  for (int kc = 0; kc < 2; ++kc) B2f[kc] = load_B_frag(sW2, 16, kc * 32, 0, lane);
  B3f = load_B_frag(sW3, 16, 0, 0, lane);

  _Float16* tile  = sX[wave];
  _Float16* tile3 = sX3[wave];
  float*    ybuf  = sY[wave];
  float*    obuf  = sO[wave];

  // FC3 K-padding (cols 16..31) is static zero: write once per wave.
#pragma unroll
  for (int r = 0; r < 8; ++r)
    tile3[(r + 8 * half) * 32 + 16 + m] = (_Float16)0.f;

  for (int t = blockIdx.x * WAVES + wave; t < ntiles; t += gridDim.x * WAVES) {
    const int p  = t * 16 + m;
    const int pc = p < n ? p : n - 1;

    // --- encoder (per lane; lanes 0-15 -> pos, 16-31 -> dir) ---
    const float* xy = (half ? dir : pos) + 2 * pc;
    const float cx = xy[0], cy = xy[1];
    const float h0 = tanhf(cx * sE[0] + cy * sE[4] + sE[8]);
    const float h1 = tanhf(cx * sE[1] + cy * sE[5] + sE[9]);
    const float h2 = tanhf(cx * sE[2] + cy * sE[6] + sE[10]);
    const float h3 = tanhf(cx * sE[3] + cy * sE[7] + sE[11]);
    const float u = sigmf(h0 * sE[12] + h1 * sE[15] + h2 * sE[18] + h3 * sE[21] + sE[24]);
    const float v = sigmf(h0 * sE[13] + h1 * sE[16] + h2 * sE[19] + h3 * sE[22] + sE[25]);

    // --- bilinear gather (L2-resident grids) ---
    const float* grid = half ? dir_grid : pos_grid;
    const float fx = u * (float)(GWN - 1);
    const float fy = v * (float)(GHN - 1);
    const int ix0 = (int)fx, iy0 = (int)fy;
    const float xf = fx - (float)ix0, yf = fy - (float)iy0;
    const int ix1 = min(ix0 + 1, GWN - 1), iy1 = min(iy0 + 1, GHN - 1);
    const float w00 = (1.f - xf) * (1.f - yf), w01 = xf * (1.f - yf);
    const float w10 = (1.f - xf) * yf,         w11 = xf * yf;
    const float4* g00 = (const float4*)(grid + (iy0 * GWN + ix0) * FNUM);
    const float4* g01 = (const float4*)(grid + (iy0 * GWN + ix1) * FNUM);
    const float4* g10 = (const float4*)(grid + (iy1 * GWN + ix0) * FNUM);
    const float4* g11 = (const float4*)(grid + (iy1 * GWN + ix1) * FNUM);
    _Float16* row = tile + m * 64 + half * 32;
#pragma unroll
    for (int q = 0; q < 8; ++q) {
      const float4 a = g00[q], b = g01[q], c = g10[q], d = g11[q];
      row[4 * q + 0] = (_Float16)(w00 * a.x + w01 * b.x + w10 * c.x + w11 * d.x);
      row[4 * q + 1] = (_Float16)(w00 * a.y + w01 * b.y + w10 * c.y + w11 * d.y);
      row[4 * q + 2] = (_Float16)(w00 * a.z + w01 * b.z + w10 * c.z + w11 * d.z);
      row[4 * q + 3] = (_Float16)(w00 * a.w + w01 * b.w + w10 * c.w + w11 * d.w);
    }

    // --- FC1: [16x64] @ [64x64], 8 WMMAs (C starts as inline 0, bias in epilogue) ---
    const v16h A0 = load_A_frag(tile, 64, 0, lane);
    const v16h A1 = load_A_frag(tile, 64, 32, lane);
    v8f acc[4];
#pragma unroll
    for (int nt = 0; nt < 4; ++nt) {
      v8f c = {};
      c = __builtin_amdgcn_wmma_f32_16x16x32_f16(false, A0, false, B1f[nt][0], (short)0, c, false, false);
      c = __builtin_amdgcn_wmma_f32_16x16x32_f16(false, A1, false, B1f[nt][1], (short)0, c, false, false);
      acc[nt] = c;
    }
#pragma unroll
    for (int nt = 0; nt < 4; ++nt) {
      const float bias = sB1[nt * 16 + m];
#pragma unroll
      for (int r = 0; r < 8; ++r)
        tile[(r + 8 * half) * 64 + nt * 16 + m] = (_Float16)leakyf(acc[nt][r] + bias);
    }

    // --- FC2: [16x64] @ [64x16], 2 WMMAs ---
    const v16h A20 = load_A_frag(tile, 64, 0, lane);
    const v16h A21 = load_A_frag(tile, 64, 32, lane);
    v8f c2 = {};
    c2 = __builtin_amdgcn_wmma_f32_16x16x32_f16(false, A20, false, B2f[0], (short)0, c2, false, false);
    c2 = __builtin_amdgcn_wmma_f32_16x16x32_f16(false, A21, false, B2f[1], (short)0, c2, false, false);
    {
      const float bias = sB2[m];
#pragma unroll
      for (int r = 0; r < 8; ++r)
        tile3[(r + 8 * half) * 32 + m] = (_Float16)leakyf(c2[r] + bias);
    }

    // --- FC3: [16x16 (K pad 32)] @ [16x8 (padded 32x16)], 1 WMMA ---
    const v16h A3 = load_A_frag(tile3, 32, 0, lane);
    v8f c3 = {};
    c3 = __builtin_amdgcn_wmma_f32_16x16x32_f16(false, A3, false, B3f, (short)0, c3, false, false);
    if (m < 8) {
      const float bias = sB3[m];
#pragma unroll
      for (int r = 0; r < 8; ++r)
        ybuf[(r + 8 * half) * 8 + m] = leakyf(c3[r] + bias);
    }

    // --- FC4 + sigmoid*255 (lanes 0..15, one point each) ---
    if (half == 0) {
      const float4* yv = (const float4*)(ybuf + m * 8);
      const float4 xa = yv[0], xb = yv[1];
      float o0 = sB4[0], o1 = sB4[1], o2 = sB4[2];
      const float xs[8] = {xa.x, xa.y, xa.z, xa.w, xb.x, xb.y, xb.z, xb.w};
#pragma unroll
      for (int k = 0; k < 8; ++k) {
        o0 += xs[k] * sW4[k * 3 + 0];
        o1 += xs[k] * sW4[k * 3 + 1];
        o2 += xs[k] * sW4[k * 3 + 2];
      }
      obuf[m * 3 + 0] = sigmf(leakyf(o0)) * 255.f;
      obuf[m * 3 + 1] = sigmf(leakyf(o1)) * 255.f;
      obuf[m * 3 + 2] = sigmf(leakyf(o2)) * 255.f;
    }

    // --- coalesced store: 48 floats per tile as 12 float4 (lanes 0..11) ---
    if (t * 16 + 15 < n) {
      if (lane < 12) {
        const float4 ov = ((const float4*)obuf)[lane];
        ((float4*)(out + (size_t)t * 48))[lane] = ov;
      }
    } else if (half == 0) {
      const int pp = t * 16 + m;
      if (pp < n) {
        out[pp * 3 + 0] = obuf[m * 3 + 0];
        out[pp * 3 + 1] = obuf[m * 3 + 1];
        out[pp * 3 + 2] = obuf[m * 3 + 2];
      }
    }
  }
}

extern "C" void kernel_launch(void* const* d_in, const int* in_sizes, int n_in,
                              void* d_out, int out_size, void* d_ws, size_t ws_size,
                              hipStream_t stream) {
  (void)n_in; (void)out_size; (void)d_ws; (void)ws_size;
  const float* pos      = (const float*)d_in[0];
  const float* dir      = (const float*)d_in[1];
  const float* pos_grid = (const float*)d_in[2];
  const float* dir_grid = (const float*)d_in[3];
  const float* ew1 = (const float*)d_in[4];
  const float* eb1 = (const float*)d_in[5];
  const float* ew2 = (const float*)d_in[6];
  const float* eb2 = (const float*)d_in[7];
  const float* w1  = (const float*)d_in[8];
  const float* b1  = (const float*)d_in[9];
  const float* w2  = (const float*)d_in[10];
  const float* b2  = (const float*)d_in[11];
  const float* w3  = (const float*)d_in[12];
  const float* b3  = (const float*)d_in[13];
  const float* w4  = (const float*)d_in[14];
  const float* b4  = (const float*)d_in[15];
  float* out = (float*)d_out;

  const int n = in_sizes[0] / 2;
  const int ntiles = (n + 15) / 16;
  int blocks = (ntiles + WAVES - 1) / WAVES;
  if (blocks > 16384) blocks = 16384;

  hipLaunchKernelGGL(gridnet_kernel, dim3(blocks), dim3(256), 0, stream,
                     pos, dir, pos_grid, dir_grid, ew1, eb1, ew2, eb2,
                     w1, b1, w2, b2, w3, b3, w4, b4, out, n, ntiles);
}
